// LoRALinear_91122026152535
// MI455X (gfx1250) — matrix operand order repaired and checked by reading, not compiled, not run on hardware
//
#include <hip/hip_runtime.h>
#include <hip/hip_bf16.h>

// ---------------------------------------------------------------------------
// LoRA linear: out = X @ W^T + SCALING * ((X @ A^T) @ B^T)
//   X: [M, K] f32, W: [N, K] f32, lora_A: [4, K] f32, lora_B: [N, 4] f32
//   M = 8192, N = 4096, K = 4096, rank = 4, SCALING = 4.0
// Strategy: bf16x3-split WMMA GEMM (v_wmma_f32_16x16x32_bf16), double-buffered
// LDS staging (one barrier per K-step), LoRA fused in epilogue from a
// precomputed xa = X @ A^T kept in workspace.
// ---------------------------------------------------------------------------

typedef __bf16 v16bf __attribute__((ext_vector_type(16)));
typedef __bf16 v8bf  __attribute__((ext_vector_type(8)));
typedef float  v8f   __attribute__((ext_vector_type(8)));

#define TILE_M 128
#define TILE_N 128
#define TILE_K 32
#define LDSTR  40   // u16 stride: 80B rows -> 16B-aligned b128 access, conflict-free columns
#define SCALING 4.0f

// ---------------- Kernel 1: xa[m, 0..3] = sum_k X[m,k] * A[r,k] -------------
__global__ __launch_bounds__(256) void lora_xa_kernel(
    const float* __restrict__ X, const float* __restrict__ A,
    float* __restrict__ xa, int K) {
  const int m = blockIdx.x;
  const int t = threadIdx.x;
  const float* xr = X + (size_t)m * K;
  float s0 = 0.f, s1 = 0.f, s2 = 0.f, s3 = 0.f;
  for (int k = t; k < K; k += 256) {
    float xv = xr[k];
    s0 += xv * A[k];
    s1 += xv * A[K + k];
    s2 += xv * A[2 * K + k];
    s3 += xv * A[3 * K + k];
  }
  __shared__ float red[4][256];
  red[0][t] = s0; red[1][t] = s1; red[2][t] = s2; red[3][t] = s3;
  __syncthreads();
  for (int off = 128; off > 0; off >>= 1) {
    if (t < off) {
      red[0][t] += red[0][t + off];
      red[1][t] += red[1][t + off];
      red[2][t] += red[2][t + off];
      red[3][t] += red[3][t + off];
    }
    __syncthreads();
  }
  if (t == 0) {
    float4 r = make_float4(red[0][0], red[1][0], red[2][0], red[3][0]);
    ((float4*)xa)[m] = r;
  }
}

// ---------------- bf16 hi/lo split helpers ---------------------------------
__device__ __forceinline__ void bf16_split(float x, unsigned short& h, unsigned short& l) {
  __bf16 hb = (__bf16)x;            // RNE to bf16
  float  hf = (float)hb;
  __bf16 lb = (__bf16)(x - hf);     // residual
  h = __builtin_bit_cast(unsigned short, hb);
  l = __builtin_bit_cast(unsigned short, lb);
}

struct StageRegs {
  float4 a[4];   // 16 f32 of the X tile
  float4 b[4];   // 16 f32 of the W tile
};

__device__ __forceinline__ void load_regs(const float* __restrict__ gA,
                                          const float* __restrict__ gB,
                                          StageRegs& r) {
  const float4* a4 = (const float4*)gA;
  const float4* b4 = (const float4*)gB;
#pragma unroll
  for (int q = 0; q < 4; ++q) r.a[q] = a4[q];
#pragma unroll
  for (int q = 0; q < 4; ++q) r.b[q] = b4[q];
}

// Split 16 f32 -> packed bf16 hi/lo, write as 2x ds_store_b128 per array.
__device__ __forceinline__ void split_store16(const float4* v,
                                              unsigned short* hiArr,
                                              unsigned short* loArr,
                                              int base /* u16 units */) {
  unsigned int h[8], l[8];
#pragma unroll
  for (int q = 0; q < 4; ++q) {
    unsigned short h0, l0, h1, l1, h2, l2, h3, l3;
    bf16_split(v[q].x, h0, l0);
    bf16_split(v[q].y, h1, l1);
    bf16_split(v[q].z, h2, l2);
    bf16_split(v[q].w, h3, l3);
    h[2 * q]     = (unsigned int)h0 | ((unsigned int)h1 << 16);
    h[2 * q + 1] = (unsigned int)h2 | ((unsigned int)h3 << 16);
    l[2 * q]     = (unsigned int)l0 | ((unsigned int)l1 << 16);
    l[2 * q + 1] = (unsigned int)l2 | ((unsigned int)l3 << 16);
  }
  uint4* ph = (uint4*)(hiArr + base);  // base*2 bytes is 16B-aligned
  uint4* pl = (uint4*)(loArr + base);
  ph[0] = make_uint4(h[0], h[1], h[2], h[3]);
  ph[1] = make_uint4(h[4], h[5], h[6], h[7]);
  pl[0] = make_uint4(l[0], l[1], l[2], l[3]);
  pl[1] = make_uint4(l[4], l[5], l[6], l[7]);
}

// A fragment (16x32 bf16, ISA 7.12.2): lane m=lane&15, half=lane>>4.
// element j -> k = (j&7) + 16*(j>>3) + 8*half  => two contiguous 8-elt chunks.
__device__ __forceinline__ v16bf ld_fragA(const unsigned short* p, int row, int half) {
  const v8bf c0 = *(const v8bf*)(p + row * LDSTR + half * 8);
  const v8bf c1 = *(const v8bf*)(p + row * LDSTR + 16 + half * 8);
  return __builtin_shufflevector(c0, c1, 0, 1, 2, 3, 4, 5, 6, 7,
                                 8, 9, 10, 11, 12, 13, 14, 15);
}

// B fragment (32x16 bf16): lane n=lane&15, half=lane>>4.
// element j -> k = (j&7) + 16*(j>>3) + 8*half  => two contiguous 8-elt chunks.
__device__ __forceinline__ v16bf ld_fragB(const unsigned short* p, int row, int half) {
  const v8bf c0 = *(const v8bf*)(p + row * LDSTR + half * 8);
  const v8bf c1 = *(const v8bf*)(p + row * LDSTR + 16 + half * 8);
  return __builtin_shufflevector(c0, c1, 0, 1, 2, 3, 4, 5, 6, 7,
                                 8, 9, 10, 11, 12, 13, 14, 15);
}

__device__ __forceinline__ v8f wmma_bf16(v16bf a, v16bf b, v8f c) {
  return __builtin_amdgcn_wmma_f32_16x16x32_bf16(
      /*neg_a=*/false, a, /*neg_b=*/false, b,
      /*c_mod=*/(short)0, c, /*reuse_a=*/false, /*reuse_b=*/false);
}

// ---------------- Kernel 2: tiled WMMA GEMM + fused LoRA epilogue -----------
__global__ __launch_bounds__(256) void lora_gemm_kernel(
    const float* __restrict__ X, const float* __restrict__ W,
    const float* __restrict__ loraB, const float* __restrict__ xa,
    float* __restrict__ out, int M, int N, int K) {
  // Double-buffered staging: 2 x (Ah, Al, Bh, Bl) = 2 x 40KB = 80KB LDS.
  __shared__ unsigned short sAh[2][TILE_M * LDSTR];
  __shared__ unsigned short sAl[2][TILE_M * LDSTR];
  __shared__ unsigned short sBh[2][TILE_N * LDSTR];
  __shared__ unsigned short sBl[2][TILE_N * LDSTR];

  const int ntiles = N / TILE_N;
  const int bn = blockIdx.x % ntiles;
  const int bm = blockIdx.x / ntiles;

  const int t = threadIdx.x;
  const int lane = t & 31;
  const int wave = t >> 5;
  const int wm = wave & 1;   // wave rows: 2 x 64
  const int wn = wave >> 1;  // wave cols: 4 x 32
  const int lane16 = lane & 15;
  const int half = lane >> 4;

  // staging assignment: each thread loads 16 f32 of each tile per K-step
  const int lrow = t >> 1;             // 0..127
  const int lcol = (t & 1) << 4;       // 0 or 16
  const int sbase = lrow * LDSTR + lcol;
  const float* gA = X + (size_t)(bm * TILE_M + lrow) * K + lcol;
  const float* gB = W + (size_t)(bn * TILE_N + lrow) * K + lcol;

  v8f acc[4][2];
#pragma unroll
  for (int i = 0; i < 4; ++i)
#pragma unroll
    for (int j = 0; j < 2; ++j) acc[i][j] = (v8f)(0.0f);

  // Prologue: stage K-step 0 into buffer 0.
  StageRegs r;
  load_regs(gA, gB, r);
  split_store16(r.a, sAh[0], sAl[0], sbase);
  split_store16(r.b, sBh[0], sBl[0], sbase);
  __syncthreads();

  for (int kt = 0; kt < K; kt += TILE_K) {
    const int cur = (kt / TILE_K) & 1;
    const int nxt = cur ^ 1;
    const bool more = (kt + TILE_K) < K;

    // Issue next stage's global loads early; latency hides under the WMMAs.
    if (more) load_regs(gA + kt + TILE_K, gB + kt + TILE_K, r);
    if (kt + 2 * TILE_K < K) {
      __builtin_prefetch(gA + kt + 2 * TILE_K, 0, 0);  // global_prefetch_b8
      __builtin_prefetch(gB + kt + 2 * TILE_K, 0, 0);
    }

    // Fragments + 24 WMMAs on the current stage.
    v16bf ah[4], al[4], bh[2], bl[2];
#pragma unroll
    for (int fn = 0; fn < 2; ++fn) {
      int row = wn * 32 + fn * 16 + lane16;
      bh[fn] = ld_fragB(sBh[cur], row, half);
      bl[fn] = ld_fragB(sBl[cur], row, half);
    }
#pragma unroll
    for (int fm = 0; fm < 4; ++fm) {
      int row = wm * 64 + fm * 16 + lane16;
      ah[fm] = ld_fragA(sAh[cur], row, half);
      al[fm] = ld_fragA(sAl[cur], row, half);
    }

#pragma unroll
    for (int fm = 0; fm < 4; ++fm) {
#pragma unroll
      for (int fn = 0; fn < 2; ++fn) {
        // bf16x3: hi*hi + lo*hi + hi*lo  (f32 accumulate)
        acc[fm][fn] = wmma_bf16(ah[fm], bh[fn], acc[fm][fn]);
        acc[fm][fn] = wmma_bf16(al[fm], bh[fn], acc[fm][fn]);
        acc[fm][fn] = wmma_bf16(ah[fm], bl[fn], acc[fm][fn]);
      }
    }

    // Convert + store next stage, then a single barrier per iteration.
    if (more) {
      split_store16(r.a, sAh[nxt], sAl[nxt], sbase);
      split_store16(r.b, sBh[nxt], sBl[nxt], sbase);
    }
    __syncthreads();
  }

  // Epilogue: out[m,n] = acc + SCALING * dot4(xa[m,:], lora_B[n,:])
  const float4* lb4 = (const float4*)loraB;  // lora_B row n is contiguous float4
  const float4* xa4 = (const float4*)xa;
#pragma unroll
  for (int fn = 0; fn < 2; ++fn) {
    int n = bn * TILE_N + wn * 32 + fn * 16 + lane16;
    float4 wb = lb4[n];
#pragma unroll
    for (int fm = 0; fm < 4; ++fm) {
      int mbase = bm * TILE_M + wm * 64 + fm * 16 + half * 8;
#pragma unroll
      for (int i = 0; i < 8; ++i) {
        int m = mbase + i;
        float4 xv = xa4[m];
        float lora = xv.x * wb.x + xv.y * wb.y + xv.z * wb.z + xv.w * wb.w;
        out[(size_t)m * N + n] = acc[fm][fn][i] + SCALING * lora;
      }
    }
  }
}

// ---------------------------------------------------------------------------
extern "C" void kernel_launch(void* const* d_in, const int* in_sizes, int n_in,
                              void* d_out, int out_size, void* d_ws, size_t ws_size,
                              hipStream_t stream) {
  const float* x      = (const float*)d_in[0];  // [B, L, K]
  const float* W      = (const float*)d_in[1];  // [N, K]
  const float* lora_A = (const float*)d_in[2];  // [4, K]
  const float* lora_B = (const float*)d_in[3];  // [N, 4]
  float* out = (float*)d_out;
  float* xa  = (float*)d_ws;                    // [M, 4] scratch

  const int K = 4096;                 // IN_F
  const int M = in_sizes[0] / K;      // B * L = 8192
  const int N = in_sizes[1] / K;      // OUT_F = 4096

  lora_xa_kernel<<<M, 256, 0, stream>>>(x, lora_A, xa, K);

  const int grid = (M / TILE_M) * (N / TILE_N);
  lora_gemm_kernel<<<grid, 256, 0, stream>>>(x, W, lora_B, xa, out, M, N, K);
}
